// TextureAttention_12936441496224
// MI455X (gfx1250) — matrix operand order repaired and checked by reading, not compiled, not run on hardware
//
#include <hip/hip_runtime.h>

typedef __attribute__((ext_vector_type(16))) _Float16 v16h;
typedef __attribute__((ext_vector_type(8)))  _Float16 v8h;
typedef __attribute__((ext_vector_type(8)))  float    v8f;
typedef __attribute__((ext_vector_type(4)))  unsigned int v4u;
typedef __attribute__((ext_vector_type(8)))  int v8i;
typedef __attribute__((ext_vector_type(4)))  int v4i;

#define BB   4
#define CC   256
#define HH   256
#define WW   256
#define HDS  64
#define NN   4096   // 64*64

#if defined(__has_builtin)
#if __has_builtin(__builtin_amdgcn_tensor_load_to_lds)
#define HAVE_TDM 1
#endif
#endif
#ifndef HAVE_TDM
#define HAVE_TDM 0
#endif

// A-matrix fragment (16-bit, 16x32): lane m holds 16 K-contiguous halves {0..15} / {16..31}
__device__ __forceinline__ v16h ld_afrag(const _Float16* p) {
  v8h lo = *(const v8h*)p;
  v8h hh = *(const v8h*)(p + 8);
  return __builtin_shufflevector(lo, hh, 0, 1, 2, 3, 4, 5, 6, 7,
                                 8, 9, 10, 11, 12, 13, 14, 15);
}
// B-matrix fragment (16-bit, 32x16): lane n holds 16 K-contiguous halves
__device__ __forceinline__ v16h ld_bfrag(const _Float16* p) {
  v8h lo = *(const v8h*)p;
  v8h hh = *(const v8h*)(p + 8);
  return __builtin_shufflevector(lo, hh, 0, 1, 2, 3, 4, 5, 6, 7,
                                 8, 9, 10, 11, 12, 13, 14, 15);
}

#if HAVE_TDM
// Issue a TDM 2D tile load (16-bit elements) global -> LDS.  D# per ISA ch.8:
// group0: [1:0]=count(1), [63:32]=lds byte addr, [120:64]=global byte addr, [127:126]=type(2)
// group1: [17:16]=data_size(1=2B), [79:48]=tensor_dim0, [111:80]=tensor_dim1,
//         [127:112]=tile_dim0, [143:128]=tile_dim1, [207:160]=tensor_dim0_stride
__device__ __forceinline__ void tdm_load_2d(unsigned lds_addr, const void* gp,
                                            unsigned tile_d0, unsigned tile_d1,
                                            unsigned tensor_d0, unsigned tensor_d1,
                                            unsigned long long stride_d0) {
  unsigned long long ga = (unsigned long long)(size_t)gp;
  v4u g0;
  g0[0] = 1u;                                   // count=1, user descriptor
  g0[1] = lds_addr;                             // LDS byte address
  g0[2] = (unsigned)ga;                         // global addr [95:64]
  g0[3] = ((unsigned)(ga >> 32) & 0x01FFFFFFu) | (2u << 30);  // addr msbs | type=2
  v8i g1;
  g1[0] = (int)(1u << 16);                      // workgroup_mask=0, data_size=2B
  g1[1] = (int)((tensor_d0 & 0xFFFFu) << 16);   // tensor_dim0 lo16
  g1[2] = (int)((tensor_d0 >> 16) | ((tensor_d1 & 0xFFFFu) << 16));
  g1[3] = (int)((tensor_d1 >> 16) | (tile_d0 << 16));
  g1[4] = (int)(tile_d1 & 0xFFFFu);             // tile_dim1, tile_dim2=0
  g1[5] = (int)(unsigned)stride_d0;             // dim0 stride lo32
  g1[6] = (int)((unsigned)(stride_d0 >> 32) & 0xFFFFu);
  g1[7] = 0;
  v4i z4 = {0, 0, 0, 0};                        // <=2D tensor: groups 2/3 unused
  v8i z8 = {0, 0, 0, 0, 0, 0, 0, 0};
  __builtin_amdgcn_tensor_load_to_lds(g0, g1, z4, z4, z8, 0);
}
#endif

// ---------------------------------------------------------------------------
// Kernel 1: x downsample (avg of 2x2 at rows/cols {4i+1,4i+2})
// ---------------------------------------------------------------------------
__global__ __launch_bounds__(256) void ta_xds_kernel(
    const float* __restrict__ x, float* __restrict__ xds, _Float16* __restrict__ vf) {
  int idx = blockIdx.x * 256 + threadIdx.x;      // < B*C*N
  int n  = idx & (NN - 1);
  int bc = idx >> 12;
  int i = n >> 6, j = n & 63;
  const float* p = x + (size_t)bc * (HH * WW) + (4 * i + 1) * WW + (4 * j + 1);
  float v = 0.25f * (p[0] + p[1] + p[WW] + p[WW + 1]);
  xds[idx] = v;
  vf[idx] = (_Float16)v;
}

// ---------------------------------------------------------------------------
// Kernel 2: pack conv weights tap-major as f16 with BN scale folded in.
// ---------------------------------------------------------------------------
__global__ __launch_bounds__(256) void ta_wprep_kernel(
    const float* __restrict__ w, const float* __restrict__ cb,
    const float* __restrict__ bs, const float* __restrict__ bb,
    const float* __restrict__ rm, const float* __restrict__ rv,
    _Float16* __restrict__ wf, float* __restrict__ betag) {
  int idx = blockIdx.x * 256 + threadIdx.x;   // < 9*256*256
  int cin  = idx & 255;
  int cout = (idx >> 8) & 255;
  int tap  = idx >> 16;
  float inv = bs[cout] * rsqrtf(rv[cout] + 1e-5f);
  wf[idx] = (_Float16)(w[((size_t)(cout * CC + cin)) * 9 + tap] * inv);
  if (idx < 256) {
    float inv2 = bs[idx] * rsqrtf(rv[idx] + 1e-5f);
    betag[idx] = (cb[idx] - rm[idx]) * inv2 + bb[idx];
  }
}

// ---------------------------------------------------------------------------
// Kernel 3: conv3x3+BN+ReLU at sampled points as 9 tap-GEMMs on WMMA,
// 2x2 relu-average via shuffles, fused per-position L2 norm -> f16 Q/K (b,n,c)
// ---------------------------------------------------------------------------
__global__ __launch_bounds__(256) void ta_conv_wmma_kernel(
    const float* __restrict__ x, const _Float16* __restrict__ wf,
    const float* __restrict__ betag, _Float16* __restrict__ qf) {
  __shared__ _Float16 xp[256 * 32];   // [pixel 0..255][cin 0..31]  (16 KB)
  __shared__ float ssq[16];
  const int t = threadIdx.x;
  const int w = t >> 5, lane = t & 31;
  const int ll = lane & 15, hi = lane >> 4;
  const int b = blockIdx.y;
  const int tile = blockIdx.x;        // 0..255
  const int ti = tile >> 4, tj = tile & 15;
  const int R0 = ti * 16, C0 = tj * 16;

  v8f acc[2][4];
#pragma unroll
  for (int mt = 0; mt < 2; ++mt)
#pragma unroll
    for (int s = 0; s < 4; ++s)
#pragma unroll
      for (int e = 0; e < 8; ++e) acc[mt][s][e] = 0.f;

  const int pc = 4 * ((ll & 7) >> 1) + 1 + (ll & 1);
  const int prpar = ll >> 3;

  const float* xb = x + (size_t)b * CC * (HH * WW);

  for (int cchunk = 0; cchunk < 8; ++cchunk) {
    const int c0 = cchunk * 32;
    __syncthreads();
#pragma unroll 4
    for (int it = 0; it < 32; ++it) {
      int lin = it * 256 + t;
      int cl = lin >> 8, px = lin & 255;
      xp[px * 32 + cl] =
          (_Float16)xb[(size_t)(c0 + cl) * (HH * WW) + (R0 + (px >> 4)) * WW + (C0 + (px & 15))];
    }
    __syncthreads();
    for (int tap = 0; tap < 9; ++tap) {
      const int dr = tap / 3 - 1, dc = tap % 3 - 1;
      v16h Aw[2];
#pragma unroll
      for (int mt = 0; mt < 2; ++mt) {
        const _Float16* wp =
            wf + ((size_t)(tap * CC + (w * 32 + mt * 16 + ll))) * CC + c0 + hi * 16;
        Aw[mt] = ld_afrag(wp);
      }
#pragma unroll
      for (int s = 0; s < 4; ++s) {
        const int pr = 4 * s + 1 + prpar;
        const int px = (pr + dr) * 16 + (pc + dc);
        v16h Bf = ld_bfrag(&xp[px * 32 + hi * 16]);
#pragma unroll
        for (int mt = 0; mt < 2; ++mt)
          acc[mt][s] = __builtin_amdgcn_wmma_f32_16x16x32_f16(false, Aw[mt], false, Bf,
                                                              (short)0, acc[mt][s],
                                                              false, false);
      }
    }
  }

  __syncthreads();
  if (t < 16) ssq[t] = 0.f;
  __syncthreads();
  const bool active = ((ll & 1) == 0) && (ll < 8);
  const int pj = (ll & 7) >> 1;
#pragma unroll
  for (int mt = 0; mt < 2; ++mt)
#pragma unroll
    for (int s = 0; s < 4; ++s)
#pragma unroll
      for (int e = 0; e < 8; ++e) {
        int cout = w * 32 + mt * 16 + e + 8 * hi;
        float y = fmaxf(acc[mt][s][e] + betag[cout], 0.f);
        float r = y + __shfl_xor(y, 1);
        r += __shfl_xor(r, 8);
        r *= 0.25f;
        acc[mt][s][e] = r;
        if (active) atomicAdd(&ssq[s * 4 + pj], r * r);
      }
  __syncthreads();
  if (active) {
#pragma unroll
    for (int s = 0; s < 4; ++s) {
      float scale = 1.f / fmaxf(sqrtf(ssq[s * 4 + pj]), 1e-12f);
      int n = (ti * 4 + s) * HDS + (tj * 4 + pj);
      _Float16* qrow = qf + ((size_t)(b * NN + n)) * CC;
#pragma unroll
      for (int mt = 0; mt < 2; ++mt)
#pragma unroll
        for (int e = 0; e < 8; ++e) {
          int cout = w * 32 + mt * 16 + e + 8 * hi;
          qrow[cout] = (_Float16)(acc[mt][s][e] * scale);
        }
    }
  }
}

// ---------------------------------------------------------------------------
// Kernel 4: flash attention, fixed-max softmax, TDM double-buffered K/V tiles.
// Block: 128 threads = 4 waves; wave 0 is the DMA leader.
// ---------------------------------------------------------------------------
__global__ __launch_bounds__(128) void ta_attn_kernel(
    const _Float16* __restrict__ qf, const _Float16* __restrict__ vf,
    const float* __restrict__ xds, const float* __restrict__ gamma,
    float* __restrict__ outds) {
  extern __shared__ float4 smem4[];
  _Float16* k_buf = (_Float16*)smem4;            // 2 x [64][256]
  _Float16* v_buf = k_buf + 2 * 64 * 256;        // 2 x [256][64]
  _Float16* p_lds = v_buf + 2 * 256 * 64;        // [4 waves][16][64]

  const int t = threadIdx.x;
  const int w = t >> 5, lane = t & 31;
  const int ll = lane & 15, hi = lane >> 4;
  const int b  = blockIdx.y;
  const int m0 = blockIdx.x * 64;
  const float g = gamma[0];

  // Q fragments resident in VGPRs (A-matrix layout)
  v16h Aq[8];
  {
    const _Float16* qrow = qf + ((size_t)(b * NN + m0 + w * 16 + ll)) * CC;
#pragma unroll
    for (int kc = 0; kc < 8; ++kc) Aq[kc] = ld_afrag(qrow + kc * 32 + hi * 16);
  }

  v8f l_acc;
#pragma unroll
  for (int e = 0; e < 8; ++e) l_acc[e] = 0.f;
  v8f O[16];
#pragma unroll
  for (int cc = 0; cc < 16; ++cc)
#pragma unroll
    for (int e = 0; e < 8; ++e) O[cc][e] = 0.f;

  _Float16* p_w = p_lds + w * 16 * 64;

#if HAVE_TDM
  const bool leader = (w == 0);
  const unsigned ka0 = (unsigned)(size_t)(void*)k_buf;
  const unsigned ka1 = (unsigned)(size_t)(void*)(k_buf + 64 * 256);
  const unsigned va0 = (unsigned)(size_t)(void*)v_buf;
  const unsigned va1 = (unsigned)(size_t)(void*)(v_buf + 256 * 64);
  if (leader) {  // prime buffer 0 with tile 0
    tdm_load_2d(ka0, qf + (size_t)(b * NN) * CC, 256, 64, 256, NN, 256);
    tdm_load_2d(va0, vf + (size_t)(b * CC) * NN, 64, 256, NN, CC, NN);
  }
#endif

  for (int iter = 0; iter < NN / 64; ++iter) {
    const int n0 = iter * 64;
    const int cur = iter & 1;
    _Float16* k_lds = k_buf + cur * (64 * 256);
    _Float16* v_lds = v_buf + cur * (256 * 64);

#if HAVE_TDM
    if (leader) {
      if (iter + 1 < NN / 64) {   // prefetch next tile into the other buffer
        const int nn = n0 + 64;
        tdm_load_2d(cur ? ka0 : ka1, qf + ((size_t)(b * NN + nn)) * CC, 256, 64, 256, NN, 256);
        tdm_load_2d(cur ? va0 : va1, vf + ((size_t)(b * CC)) * NN + nn, 64, 256, NN, CC, NN);
        __builtin_amdgcn_s_wait_tensorcnt(2);   // in-order: current tile's 2 ops done
      } else {
        __builtin_amdgcn_s_wait_tensorcnt(0);
      }
    }
#else
    { // synchronous fallback: K tile [row][256], V tile [ch][64]
      const float4* src = (const float4*)(qf + ((size_t)(b * NN + n0)) * CC);
      float4* dst = (float4*)k_lds;
#pragma unroll 4
      for (int it = 0; it < 32; ++it) dst[it * 128 + t] = src[it * 128 + t];
      float4* dstv = (float4*)v_lds;
#pragma unroll 4
      for (int it = 0; it < 16; ++it) {
        int lin = it * 128 + t;
        int ch = lin >> 3, f4 = lin & 7;
        dstv[lin] = ((const float4*)(vf + ((size_t)(b * CC + ch)) * NN + n0))[f4];
      }
    }
#endif
    __syncthreads();   // tile `cur` visible to all waves

    // ---- S = Q*K^T (4 subtiles), exp, P -> LDS ----
#pragma unroll
    for (int s = 0; s < 4; ++s) {
      v8f S;
#pragma unroll
      for (int e = 0; e < 8; ++e) S[e] = 0.f;
      const _Float16* kp = &k_lds[(s * 16 + ll) * 256];
      v16h Bc = ld_bfrag(kp + hi * 16);
#pragma unroll
      for (int kc = 0; kc < 8; ++kc) {
        v16h Bn = Bc;
        if (kc < 7) Bn = ld_bfrag(kp + (kc + 1) * 32 + hi * 16);
        S = __builtin_amdgcn_wmma_f32_16x16x32_f16(false, Aq[kc], false, Bc,
                                                   (short)0, S, false, false);
        Bc = Bn;
      }
#pragma unroll
      for (int e = 0; e < 8; ++e) {
        float pe = __expf(2.f * S[e] - 2.f);   // score/T with fixed max (cos<=1, T=0.5)
        l_acc[e] += pe;
        p_w[(e + 8 * hi) * 64 + s * 16 + ll] = (_Float16)pe;
      }
    }

    // ---- O += P * V ----
#pragma unroll
    for (int kc2 = 0; kc2 < 2; ++kc2) {
      v16h Ap = ld_afrag(&p_w[ll * 64 + kc2 * 32 + hi * 16]);
      v16h Bc = ld_bfrag(&v_lds[ll * 64 + kc2 * 32 + hi * 16]);
#pragma unroll
      for (int cc = 0; cc < 16; ++cc) {
        v16h Bn = Bc;
        if (cc < 15) Bn = ld_bfrag(&v_lds[((cc + 1) * 16 + ll) * 64 + kc2 * 32 + hi * 16]);
        O[cc] = __builtin_amdgcn_wmma_f32_16x16x32_f16(false, Ap, false, Bc,
                                                       (short)0, O[cc], false, false);
        Bc = Bn;
      }
    }
    __syncthreads();   // all readers done before this buffer is re-targeted
  }

  float invl[8];
#pragma unroll
  for (int e = 0; e < 8; ++e) {
    float sv = l_acc[e];
    sv += __shfl_xor(sv, 1);
    sv += __shfl_xor(sv, 2);
    sv += __shfl_xor(sv, 4);
    sv += __shfl_xor(sv, 8);
    invl[e] = 1.f / fmaxf(sv, 1e-20f);
  }

#pragma unroll
  for (int cc = 0; cc < 16; ++cc) {
    int ch = cc * 16 + ll;
#pragma unroll
    for (int e = 0; e < 8; ++e) {
      int m = m0 + w * 16 + e + 8 * hi;
      size_t idx = ((size_t)(b * CC + ch)) * NN + m;
      outds[idx] = xds[idx] + g * (O[cc][e] * invl[e]);
    }
  }
}

// ---------------------------------------------------------------------------
// Kernel 5: bilinear upsample 64x64 -> 256x256 (align_corners=False, clamp)
// ---------------------------------------------------------------------------
__global__ __launch_bounds__(256) void ta_upsample_kernel(
    const float* __restrict__ outds, float* __restrict__ out) {
  int idx = blockIdx.x * 256 + threadIdx.x;   // < B*C*H*W
  int J = idx & 255;
  int I = (idx >> 8) & 255;
  int bc = idx >> 16;
  float sI = I * 0.25f - 0.375f;
  float sJ = J * 0.25f - 0.375f;
  float fI = floorf(sI), fJ = floorf(sJ);
  float wi = sI - fI, wj = sJ - fJ;
  int i0 = (int)fI, j0 = (int)fJ;
  int i0c = max(i0, 0), i1c = min(i0 + 1, 63);
  int j0c = max(j0, 0), j1c = min(j0 + 1, 63);
  const float* p = outds + (size_t)bc * NN;
  float v00 = p[i0c * 64 + j0c], v01 = p[i0c * 64 + j1c];
  float v10 = p[i1c * 64 + j0c], v11 = p[i1c * 64 + j1c];
  float v0 = v00 + wj * (v01 - v00);
  float v1 = v10 + wj * (v11 - v10);
  out[idx] = v0 + wi * (v1 - v0);
}

// ---------------------------------------------------------------------------
extern "C" void kernel_launch(void* const* d_in, const int* in_sizes, int n_in,
                              void* d_out, int out_size, void* d_ws, size_t ws_size,
                              hipStream_t stream) {
  const float* x  = (const float*)d_in[0];
  const float* cw = (const float*)d_in[1];
  const float* cb = (const float*)d_in[2];
  const float* bs = (const float*)d_in[3];
  const float* bbias = (const float*)d_in[4];
  const float* rm = (const float*)d_in[5];
  const float* rv = (const float*)d_in[6];
  const float* gm = (const float*)d_in[7];

  char* ws = (char*)d_ws;
  float*    xds   = (float*)ws;                                 // 16 MB
  float*    outds = (float*)(ws + ((size_t)16 << 20));          // 16 MB
  _Float16* qf    = (_Float16*)(ws + ((size_t)32 << 20));       //  8 MB
  _Float16* vf    = (_Float16*)(ws + ((size_t)40 << 20));       //  8 MB
  _Float16* wf    = (_Float16*)(ws + ((size_t)48 << 20));       // 1.2 MB
  float*    betag = (float*)(ws + ((size_t)50 << 20));          //  1 KB

  ta_xds_kernel<<<dim3((BB * CC * NN) / 256), 256, 0, stream>>>(x, xds, vf);
  ta_wprep_kernel<<<dim3((9 * CC * CC) / 256), 256, 0, stream>>>(cw, cb, bs, bbias, rm, rv, wf, betag);
  ta_conv_wmma_kernel<<<dim3(256, BB), 256, 0, stream>>>(x, wf, betag, qf);

  // 2x K tile + 2x V tile + P scratch = (2*16384 + 2*16384 + 4096) halves = 136 KB
  size_t shmem = (size_t)(2 * 64 * 256 + 2 * 256 * 64 + 4 * 16 * 64) * sizeof(_Float16);
  ta_attn_kernel<<<dim3(NN / 64, BB), 128, shmem, stream>>>(qf, vf, xds, gm, outds);

  ta_upsample_kernel<<<dim3((BB * CC * HH * WW) / 256), 256, 0, stream>>>(outds, (float*)d_out);
}
